// doubleAttenCombineEncoderRNN_18519898980446
// MI455X (gfx1250) — matrix operand order repaired and checked by reading, not compiled, not run on hardware
//
#include <hip/hip_runtime.h>
#include <cmath>

typedef __attribute__((ext_vector_type(16))) _Float16 v16h;
typedef __attribute__((ext_vector_type(8)))  _Float16 v8h;
typedef __attribute__((ext_vector_type(8)))  float    v8f;

#define HID   1024   // H
#define K2    2048   // 2H
#define LSPAN 30     // attention span
#define BM    16     // batch rows per workgroup
#define NTHR  256    // 8 waves (wave32)
#define BATCH 1024

// ---- WMMA tile loaders (contiguous k order, same in A and B) -------------------

// A-matrix 16x32 f16 tile: lane l -> row m=l&15, half p -> k = 16*(l>>4) + p
// => two contiguous 8-half (16B) LDS reads at col and col+8.
__device__ __forceinline__ v16h load_a_tile(const _Float16* As, int m, int col) {
  const _Float16* p = As + m * K2 + col;
  union { v16h v; v8h h[2]; } u;
  u.h[0] = *(const v8h*)(p);
  u.h[1] = *(const v8h*)(p + 8);
  return u.v;
}

// B-matrix 32x16 f16 tile: lane l -> col n=l&15, half p -> k = 16*(l>>4) + p.
// f32 weights: 16 consecutive floats, convert (8 v_cvt_pk).
__device__ __forceinline__ v16h load_b_tile(const float* w) {
  const float4* p = (const float4*)w;
  float4 f0 = p[0], f1 = p[1], f2 = p[2], f3 = p[3];
  v16h b;
  b[0]=(_Float16)f0.x;  b[1]=(_Float16)f0.y;  b[2]=(_Float16)f0.z;  b[3]=(_Float16)f0.w;
  b[4]=(_Float16)f1.x;  b[5]=(_Float16)f1.y;  b[6]=(_Float16)f1.z;  b[7]=(_Float16)f1.w;
  b[8]=(_Float16)f2.x;  b[9]=(_Float16)f2.y;  b[10]=(_Float16)f2.z; b[11]=(_Float16)f2.w;
  b[12]=(_Float16)f3.x; b[13]=(_Float16)f3.y; b[14]=(_Float16)f3.z; b[15]=(_Float16)f3.w;
  return b;
}

// f16 weights (pre-converted in ws): 16 consecutive halves, two 16B loads, no cvt.
__device__ __forceinline__ v16h load_b_tile(const _Float16* w) {
  union { v16h v; v8h h[2]; } u;
  u.h[0] = *(const v8h*)(w);
  u.h[1] = *(const v8h*)(w + 8);
  return u.v;
}

// Accumulate D += A[m, acol0 ...] * W_row^T for one 16x16 tile.
template <typename WT>
__device__ __forceinline__ v8f gemm_accum(v8f acc, const _Float16* As, int m, int g,
                                          int acol0, const WT* wrow, int ksteps) {
#pragma unroll 4
  for (int ks = 0; ks < ksteps; ++ks) {
    int kb = ks * 32;
    v16h a = load_a_tile(As, m, acol0 + kb + 16 * g);
    v16h b = load_b_tile(wrow + kb + 16 * g);
    acc = __builtin_amdgcn_wmma_f32_16x16x32_f16(false, a, false, b, (short)0, acc,
                                                 false, false);
  }
  return acc;
}

// Stage 16 rows x 2048 f16 from global workspace into LDS (coalesced 16B chunks).
__device__ __forceinline__ void stage_rows_f16(_Float16* As, const _Float16* src,
                                               int b0, int tid) {
  for (int i = tid; i < BM * K2 / 8; i += NTHR) {
    int mm = i >> 8, c8 = (i & 255) * 8;
    *(v8h*)(As + mm * K2 + c8) = *(const v8h*)(src + (size_t)(b0 + mm) * K2 + c8);
  }
}

// ===================== weight downconvert (once per call) ==================

__global__ void __launch_bounds__(NTHR)
k_cvt16(const float* __restrict__ src, _Float16* __restrict__ dst) {
  int i = (blockIdx.x * NTHR + threadIdx.x) * 8;
  float4 f0 = *(const float4*)(src + i);
  float4 f1 = *(const float4*)(src + i + 4);
  v8h o;
  o[0]=(_Float16)f0.x; o[1]=(_Float16)f0.y; o[2]=(_Float16)f0.z; o[3]=(_Float16)f0.w;
  o[4]=(_Float16)f1.x; o[5]=(_Float16)f1.y; o[6]=(_Float16)f1.z; o[7]=(_Float16)f1.w;
  *(v8h*)(dst + i) = o;
}

// ===================== multi-kernel pipeline ===============================

// K1: emb = relu([seq1|seq2] @ Wc^T + bc) -> A2[:,0:1024]   grid (64, 8)
template <typename WT>
__global__ void __launch_bounds__(NTHR)
k_emb(const float* __restrict__ seq1, const float* __restrict__ seq2,
      const WT* __restrict__ Wc, const float* __restrict__ bc,
      _Float16* __restrict__ A2)
{
  extern __shared__ char smem[];
  _Float16* As = (_Float16*)smem;
  const int tid = threadIdx.x, lane = tid & 31, wave = tid >> 5;
  const int ln = lane & 15, g = lane >> 4, m = ln;
  const int b0 = blockIdx.x * BM;

  for (int i = tid; i < BM * K2; i += NTHR) {
    int mm = i >> 11, c = i & (K2 - 1);
    float v = (c < HID) ? seq1[(size_t)(b0 + mm) * HID + c]
                        : seq2[(size_t)(b0 + mm) * HID + (c - HID)];
    As[i] = (_Float16)v;
  }
  __syncthreads();

  int n = blockIdx.y * 128 + wave * 16 + ln;
  v8f z = {};
  v8f acc = gemm_accum(z, As, m, g, 0, Wc + (size_t)n * K2, 64);
  float bias = bc[n];
#pragma unroll
  for (int r = 0; r < 8; ++r)
    A2[(size_t)(b0 + r + 8 * g) * K2 + n] = (_Float16)fmaxf(acc[r] + bias, 0.f);
}

// K2: attention logits+softmax+weighted encoder sum -> A2[:,1024:], A3[:,1024:]=f16(h)
// grid (64, 4): each block owns a 256-column slice of the encoder stream.
__global__ void __launch_bounds__(NTHR)
k_attn(_Float16* __restrict__ A2, _Float16* __restrict__ A3,
       const float* __restrict__ hidden, const float* __restrict__ enc,
       const float* __restrict__ Wattn, const float* __restrict__ battn)
{
  __shared__ float lg[BM * LSPAN];
  __shared__ float wsm[BM * LSPAN];
  const int tid = threadIdx.x;
  const int b0 = blockIdx.x * BM;
  const int c0 = blockIdx.y * 256;

  for (int q = tid; q < BM * LSPAN; q += NTHR) {
    int mq = q / LSPAN, j = q - mq * LSPAN;
    const float*    wr = Wattn + (size_t)j * K2;
    const _Float16* ar = A2 + (size_t)(b0 + mq) * K2;       // emb (f16)
    const float*    hr = hidden + (size_t)(b0 + mq) * HID;  // h (f32)
    float s = 0.f;
    for (int k = 0; k < HID; k += 8) {
      v8h a = *(const v8h*)(ar + k);
      float4 w0 = *(const float4*)(wr + k),       w1 = *(const float4*)(wr + k + 4);
      float4 u0 = *(const float4*)(wr + HID + k), u1 = *(const float4*)(wr + HID + k + 4);
      float4 h0 = *(const float4*)(hr + k),       h1 = *(const float4*)(hr + k + 4);
      s += (float)a[0]*w0.x + (float)a[1]*w0.y + (float)a[2]*w0.z + (float)a[3]*w0.w
         + (float)a[4]*w1.x + (float)a[5]*w1.y + (float)a[6]*w1.z + (float)a[7]*w1.w
         + h0.x*u0.x + h0.y*u0.y + h0.z*u0.z + h0.w*u0.w
         + h1.x*u1.x + h1.y*u1.y + h1.z*u1.z + h1.w*u1.w;
    }
    lg[q] = s + battn[j];
  }
  __syncthreads();
  if (tid < BM) {
    float mx = -1e30f;
    for (int j = 0; j < LSPAN; ++j) mx = fmaxf(mx, lg[tid * LSPAN + j]);
    float sum = 0.f;
    for (int j = 0; j < LSPAN; ++j) {
      float e = __expf(lg[tid * LSPAN + j] - mx);
      wsm[tid * LSPAN + j] = e; sum += e;
    }
    float inv = 1.f / sum;
    for (int j = 0; j < LSPAN; ++j) wsm[tid * LSPAN + j] *= inv;
  }
  __syncthreads();

  int col = c0 + tid;
  for (int mm = 0; mm < BM; ++mm) {
    size_t row = (size_t)(b0 + mm);
    const float* eb = enc + row * LSPAN * HID + col;
    float a = 0.f;
#pragma unroll 6
    for (int j = 0; j < LSPAN; ++j) a += wsm[mm * LSPAN + j] * eb[(size_t)j * HID];
    A2[row * K2 + HID + col] = (_Float16)a;
    A3[row * K2 + HID + col] = (_Float16)hidden[row * HID + col];
  }
}

// K3: out = relu(A2 @ Wcomb^T + bcomb) -> A3[:,0:1024]   grid (64, 8)
template <typename WT>
__global__ void __launch_bounds__(NTHR)
k_out(const _Float16* __restrict__ A2, const WT* __restrict__ Wcomb,
      const float* __restrict__ bcomb, _Float16* __restrict__ A3)
{
  extern __shared__ char smem[];
  _Float16* As = (_Float16*)smem;
  const int tid = threadIdx.x, lane = tid & 31, wave = tid >> 5;
  const int ln = lane & 15, g = lane >> 4, m = ln;
  const int b0 = blockIdx.x * BM;

  stage_rows_f16(As, A2, b0, tid);
  __syncthreads();

  int n = blockIdx.y * 128 + wave * 16 + ln;
  v8f z = {};
  v8f acc = gemm_accum(z, As, m, g, 0, Wcomb + (size_t)n * K2, 64);
  float bias = bcomb[n];
#pragma unroll
  for (int r = 0; r < 8; ++r)
    A3[(size_t)(b0 + r + 8 * g) * K2 + n] = (_Float16)fmaxf(acc[r] + bias, 0.f);
}

// K4: rz = sigmoid(out@Wih_rz^T + h@Whh_rz^T + b)   grid (64, 8), 2 tiles/wave
template <typename WT>
__global__ void __launch_bounds__(NTHR)
k_rz(const _Float16* __restrict__ A3, const WT* __restrict__ W_ih,
     const float* __restrict__ b_ih, const WT* __restrict__ W_hh,
     const float* __restrict__ b_hh, _Float16* __restrict__ rzw)
{
  extern __shared__ char smem[];
  _Float16* As = (_Float16*)smem;
  const int tid = threadIdx.x, lane = tid & 31, wave = tid >> 5;
  const int ln = lane & 15, g = lane >> 4, m = ln;
  const int b0 = blockIdx.x * BM;

  stage_rows_f16(As, A3, b0, tid);
  __syncthreads();

#pragma unroll
  for (int t = 0; t < 2; ++t) {
    int n = blockIdx.y * 256 + wave * 32 + t * 16 + ln;   // 0..2047
    v8f z = {};
    v8f s = gemm_accum(z, As, m, g, 0,   W_ih + (size_t)n * HID, 32);
    s     = gemm_accum(s, As, m, g, HID, W_hh + (size_t)n * HID, 32);
    float bias = b_ih[n] + b_hh[n];
#pragma unroll
    for (int r = 0; r < 8; ++r) {
      float x = s[r] + bias;
      rzw[(size_t)(b0 + r + 8 * g) * K2 + n] = (_Float16)(1.f / (1.f + __expf(-x)));
    }
  }
}

// K5: n-gate + GRU blend -> out (both copies)   grid (64, 8)
template <typename WT>
__global__ void __launch_bounds__(NTHR)
k_ngate(const _Float16* __restrict__ A3, const WT* __restrict__ W_ih,
        const float* __restrict__ b_ih, const WT* __restrict__ W_hh,
        const float* __restrict__ b_hh, const _Float16* __restrict__ rzw,
        const float* __restrict__ hidden, float* __restrict__ out)
{
  extern __shared__ char smem[];
  _Float16* As = (_Float16*)smem;
  const int tid = threadIdx.x, lane = tid & 31, wave = tid >> 5;
  const int ln = lane & 15, g = lane >> 4, m = ln;
  const int b0 = blockIdx.x * BM;

  stage_rows_f16(As, A3, b0, tid);
  __syncthreads();

  int n2 = blockIdx.y * 128 + wave * 16 + ln;             // 0..1023
  v8f zi = {}, zh = {};
  v8f si = gemm_accum(zi, As, m, g, 0,   W_ih + (size_t)(2 * HID + n2) * HID, 32);
  v8f sh = gemm_accum(zh, As, m, g, HID, W_hh + (size_t)(2 * HID + n2) * HID, 32);
  float bi = b_ih[2 * HID + n2];
  float bh = b_hh[2 * HID + n2];
#pragma unroll
  for (int r = 0; r < 8; ++r) {
    size_t row = (size_t)(b0 + r + 8 * g);
    float rr = (float)rzw[row * K2 + n2];
    float zz = (float)rzw[row * K2 + HID + n2];
    float nn = tanhf((si[r] + bi) + rr * (sh[r] + bh));
    float hv = hidden[row * HID + n2];
    float o  = (1.f - zz) * nn + zz * hv;
    out[row * HID + n2] = o;
    out[(size_t)BATCH * HID + row * HID + n2] = o;
  }
}

// ===================== fallback: fully fused single kernel =================

__global__ void __launch_bounds__(NTHR)
fused_decoder_step(const float* __restrict__ seq1,  const float* __restrict__ seq2,
                   const float* __restrict__ hidden,const float* __restrict__ enc,
                   const float* __restrict__ Wc,    const float* __restrict__ bc,
                   const float* __restrict__ Wattn, const float* __restrict__ battn,
                   const float* __restrict__ Wcomb, const float* __restrict__ bcomb,
                   const float* __restrict__ W_ih,  const float* __restrict__ b_ih,
                   const float* __restrict__ W_hh,  const float* __restrict__ b_hh,
                   float* __restrict__ out)
{
  extern __shared__ char smem[];
  _Float16* As = (_Float16*)smem;               // [16][2048]
  _Float16* rz = (_Float16*)(smem + 65536);     // [16][2048]
  float*    lg = (float*)(smem + 131072);
  float*   wsm = (float*)(smem + 133120);

  const int tid = threadIdx.x, lane = tid & 31, wave = tid >> 5;
  const int ln = lane & 15, g = lane >> 4, m = ln;
  const int b0 = blockIdx.x * BM;

  for (int i = tid; i < BM * K2; i += NTHR) {
    int mm = i >> 11, c = i & (K2 - 1);
    float v = (c < HID) ? seq1[(size_t)(b0 + mm) * HID + c]
                        : seq2[(size_t)(b0 + mm) * HID + (c - HID)];
    As[i] = (_Float16)v;
  }
  __syncthreads();

  v8f acc[8];
#pragma unroll
  for (int t = 0; t < 8; ++t) {
    int n = (wave * 8 + t) * 16 + ln;
    v8f z = {};
    acc[t] = gemm_accum(z, As, m, g, 0, Wc + (size_t)n * K2, 64);
  }
  __syncthreads();
#pragma unroll
  for (int t = 0; t < 8; ++t) {
    int n = (wave * 8 + t) * 16 + ln;
    float bias = bc[n];
#pragma unroll
    for (int r = 0; r < 8; ++r)
      As[(r + 8 * g) * K2 + n] = (_Float16)fmaxf(acc[t][r] + bias, 0.f);
  }
  for (int i = tid; i < BM * HID; i += NTHR) {
    int mm = i >> 10, c = i & (HID - 1);
    As[mm * K2 + HID + c] = (_Float16)hidden[(size_t)(b0 + mm) * HID + c];
  }
  __syncthreads();

  for (int q = tid; q < BM * LSPAN; q += NTHR) {
    int mq = q / LSPAN, j = q - mq * LSPAN;
    const float*    wr = Wattn + (size_t)j * K2;
    const _Float16* ar = As + mq * K2;
    float s = 0.f;
    for (int k = 0; k < K2; k += 8) {
      v8h a = *(const v8h*)(ar + k);
      float4 w0 = *(const float4*)(wr + k);
      float4 w1 = *(const float4*)(wr + k + 4);
      s += (float)a[0]*w0.x + (float)a[1]*w0.y + (float)a[2]*w0.z + (float)a[3]*w0.w
         + (float)a[4]*w1.x + (float)a[5]*w1.y + (float)a[6]*w1.z + (float)a[7]*w1.w;
    }
    lg[q] = s + battn[j];
  }
  __syncthreads();
  if (tid < BM) {
    float mx = -1e30f;
    for (int j = 0; j < LSPAN; ++j) mx = fmaxf(mx, lg[tid * LSPAN + j]);
    float sum = 0.f;
    for (int j = 0; j < LSPAN; ++j) {
      float e = __expf(lg[tid * LSPAN + j] - mx);
      wsm[tid * LSPAN + j] = e; sum += e;
    }
    float inv = 1.f / sum;
    for (int j = 0; j < LSPAN; ++j) wsm[tid * LSPAN + j] *= inv;
  }
  __syncthreads();

  {
    int c0 = tid * 4;
    for (int mm = 0; mm < BM; ++mm) {
      const float* eb = enc + (size_t)(b0 + mm) * LSPAN * HID + c0;
      float ax = 0.f, ay = 0.f, az = 0.f, aw = 0.f;
#pragma unroll 5
      for (int j = 0; j < LSPAN; ++j) {
        float wj = wsm[mm * LSPAN + j];
        float4 e = *(const float4*)(eb + (size_t)j * HID);
        ax += wj * e.x; ay += wj * e.y; az += wj * e.z; aw += wj * e.w;
      }
      _Float16* d = As + mm * K2 + HID + c0;
      d[0]=(_Float16)ax; d[1]=(_Float16)ay; d[2]=(_Float16)az; d[3]=(_Float16)aw;
    }
  }
  __syncthreads();

#pragma unroll
  for (int t = 0; t < 8; ++t) {
    int n = (wave * 8 + t) * 16 + ln;
    v8f z = {};
    acc[t] = gemm_accum(z, As, m, g, 0, Wcomb + (size_t)n * K2, 64);
  }
  __syncthreads();
#pragma unroll
  for (int t = 0; t < 8; ++t) {
    int n = (wave * 8 + t) * 16 + ln;
    float bias = bcomb[n];
#pragma unroll
    for (int r = 0; r < 8; ++r)
      As[(r + 8 * g) * K2 + n] = (_Float16)fmaxf(acc[t][r] + bias, 0.f);
  }
  for (int i = tid; i < BM * HID; i += NTHR) {
    int mm = i >> 10, c = i & (HID - 1);
    As[mm * K2 + HID + c] = (_Float16)hidden[(size_t)(b0 + mm) * HID + c];
  }
  __syncthreads();

#pragma unroll 2
  for (int t = 0; t < 16; ++t) {
    int n = (wave * 16 + t) * 16 + ln;
    v8f z = {};
    v8f s = gemm_accum(z, As, m, g, 0,   W_ih + (size_t)n * HID, 32);
    s     = gemm_accum(s, As, m, g, HID, W_hh + (size_t)n * HID, 32);
    float bias = b_ih[n] + b_hh[n];
#pragma unroll
    for (int r = 0; r < 8; ++r) {
      float x = s[r] + bias;
      rz[(r + 8 * g) * K2 + n] = (_Float16)(1.f / (1.f + __expf(-x)));
    }
  }
  __syncthreads();

#pragma unroll
  for (int t = 0; t < 8; ++t) {
    int n2 = (wave * 8 + t) * 16 + ln;
    v8f zi = {}, zh = {};
    v8f si = gemm_accum(zi, As, m, g, 0,   W_ih + (size_t)(2 * HID + n2) * HID, 32);
    v8f sh = gemm_accum(zh, As, m, g, HID, W_hh + (size_t)(2 * HID + n2) * HID, 32);
    float bi = b_ih[2 * HID + n2];
    float bh = b_hh[2 * HID + n2];
#pragma unroll
    for (int r = 0; r < 8; ++r) {
      int mm = r + 8 * g;
      size_t row = (size_t)(b0 + mm);
      float rr = (float)rz[mm * K2 + n2];
      float zz = (float)rz[mm * K2 + HID + n2];
      float nn = tanhf((si[r] + bi) + rr * (sh[r] + bh));
      float hv = hidden[row * HID + n2];
      float o  = (1.f - zz) * nn + zz * hv;
      out[row * HID + n2] = o;
      out[(size_t)BATCH * HID + row * HID + n2] = o;
    }
  }
}

// ===========================================================================

extern "C" void kernel_launch(void* const* d_in, const int* in_sizes, int n_in,
                              void* d_out, int out_size, void* d_ws, size_t ws_size,
                              hipStream_t stream) {
  (void)in_sizes; (void)n_in; (void)out_size;
  const float* seq1  = (const float*)d_in[0];
  const float* seq2  = (const float*)d_in[1];
  const float* hid   = (const float*)d_in[2];
  const float* enc   = (const float*)d_in[3];
  const float* Wc    = (const float*)d_in[4];
  const float* bc    = (const float*)d_in[5];
  const float* Wattn = (const float*)d_in[6];
  const float* battn = (const float*)d_in[7];
  const float* Wcomb = (const float*)d_in[8];
  const float* bcomb = (const float*)d_in[9];
  const float* W_ih  = (const float*)d_in[10];
  const float* b_ih  = (const float*)d_in[11];
  const float* W_hh  = (const float*)d_in[12];
  const float* b_hh  = (const float*)d_in[13];
  float* out = (float*)d_out;

  const size_t nAct  = (size_t)BATCH * K2;          // 2M halves per activation buf
  const size_t nW2H  = (size_t)HID * K2;            // Wc / Wcomb elements (2M)
  const size_t n3HH  = (size_t)3 * HID * HID;       // W_ih / W_hh elements (3M)
  const size_t actBytes = 3 * nAct * sizeof(_Float16);                  // 12 MB
  const size_t wBytes   = (2 * nW2H + 2 * n3HH) * sizeof(_Float16);     // 20 MB

  const dim3 G64_8(BATCH / BM, 8), G64_4(BATCH / BM, 4), B256(NTHR);

  if (ws_size >= actBytes + wBytes) {
    // tier A: activations + pre-converted f16 weights in ws (zero cvt in GEMM loops)
    _Float16* A2   = (_Float16*)d_ws;
    _Float16* A3   = A2 + nAct;
    _Float16* rzw  = A3 + nAct;
    _Float16* Wc16 = rzw + nAct;
    _Float16* Wb16 = Wc16 + nW2H;
    _Float16* Wi16 = Wb16 + nW2H;
    _Float16* Wh16 = Wi16 + n3HH;
    k_cvt16<<<dim3(nW2H / (8 * NTHR)), B256, 0, stream>>>(Wc,    Wc16);
    k_cvt16<<<dim3(nW2H / (8 * NTHR)), B256, 0, stream>>>(Wcomb, Wb16);
    k_cvt16<<<dim3(n3HH / (8 * NTHR)), B256, 0, stream>>>(W_ih,  Wi16);
    k_cvt16<<<dim3(n3HH / (8 * NTHR)), B256, 0, stream>>>(W_hh,  Wh16);
    k_emb<_Float16>  <<<G64_8, B256, 65536, stream>>>(seq1, seq2, Wc16, bc, A2);
    k_attn           <<<G64_4, B256, 0,     stream>>>(A2, A3, hid, enc, Wattn, battn);
    k_out<_Float16>  <<<G64_8, B256, 65536, stream>>>(A2, Wb16, bcomb, A3);
    k_rz<_Float16>   <<<G64_8, B256, 65536, stream>>>(A3, Wi16, b_ih, Wh16, b_hh, rzw);
    k_ngate<_Float16><<<G64_8, B256, 65536, stream>>>(A3, Wi16, b_ih, Wh16, b_hh,
                                                      rzw, hid, out);
  } else if (ws_size >= actBytes) {
    // tier B: activations in ws, on-the-fly f32->f16 weight conversion
    _Float16* A2  = (_Float16*)d_ws;
    _Float16* A3  = A2 + nAct;
    _Float16* rzw = A3 + nAct;
    k_emb<float>  <<<G64_8, B256, 65536, stream>>>(seq1, seq2, Wc, bc, A2);
    k_attn        <<<G64_4, B256, 0,     stream>>>(A2, A3, hid, enc, Wattn, battn);
    k_out<float>  <<<G64_8, B256, 65536, stream>>>(A2, Wcomb, bcomb, A3);
    k_rz<float>   <<<G64_8, B256, 65536, stream>>>(A3, W_ih, b_ih, W_hh, b_hh, rzw);
    k_ngate<float><<<G64_8, B256, 65536, stream>>>(A3, W_ih, b_ih, W_hh, b_hh,
                                                   rzw, hid, out);
  } else {
    // tier C: fully fused, zero workspace
    constexpr size_t SMEM = 65536 + 65536 + 2048 + 2048;  // 132 KB (< 320 KB WGP)
    fused_decoder_step<<<dim3(BATCH / BM), dim3(NTHR), SMEM, stream>>>(
        seq1, seq2, hid, enc, Wc, bc, Wattn, battn, Wcomb, bcomb,
        W_ih, b_ih, W_hh, b_hh, out);
  }
}
